// NonLocalBlock_16037407883930
// MI455X (gfx1250) — matrix operand order repaired and checked by reading, not compiled, not run on hardware
//
#include <hip/hip_runtime.h>
#include <hip/hip_bf16.h>

typedef unsigned short u16b;
typedef __attribute__((ext_vector_type(8)))  u16b   v8us;
typedef __attribute__((ext_vector_type(16))) u16b   v16us;
typedef __attribute__((ext_vector_type(16))) __bf16 v16bf;
typedef __attribute__((ext_vector_type(8)))  float  v8f;

#define BATCH 32
#define C_IN  256
#define CI    128
#define NSP   4096  // H*W

// ---------- helpers ----------

__device__ __forceinline__ u16b f2bf(float f) {
  unsigned int u = __float_as_uint(f);
  u += 0x7FFFu + ((u >> 16) & 1u);   // round-to-nearest-even
  return (u16b)(u >> 16);
}

// A-fragment: element e<8  -> K = half*16 + e         (16B chunk 0)
//             element e>=8 -> K = half*16 + e (16B chunk 1)
__device__ __forceinline__ v16bf frag_cat(const u16b* lo, const u16b* hi) {
  v8us a = *(const v8us*)lo;
  v8us b = *(const v8us*)hi;
  v16us c = __builtin_shufflevector(a, b, 0,1,2,3,4,5,6,7,8,9,10,11,12,13,14,15);
  return __builtin_bit_cast(v16bf, c);
}
// B-fragment: element e -> K = half*16 + e (one contiguous 32B run)
__device__ __forceinline__ v16bf frag_ld(const u16b* p) {
  return __builtin_bit_cast(v16bf, *(const v16us*)p);
}
__device__ __forceinline__ v8f wmma_bf16(v16bf a, v16bf b, v8f c) {
  return __builtin_amdgcn_wmma_f32_16x16x32_bf16(false, a, false, b, (short)0, c,
                                                 false, false);
}
__device__ __forceinline__ v8f zero8() {
  v8f z;
#pragma unroll
  for (int i = 0; i < 8; ++i) z[i] = 0.0f;
  return z;
}

// ---------- kernel 0: pack weights f32 -> bf16 fragment layout ----------
// layout per matrix: [mtile][kstep][lane(32)][e(16)]
__global__ __launch_bounds__(256) void pack_weights_kernel(
    const float* __restrict__ tw, const float* __restrict__ pw,
    const float* __restrict__ gw, const float* __restrict__ ow,
    u16b* __restrict__ wpack) {
  int idx  = blockIdx.x * 256 + threadIdx.x;  // 0..131071
  int m    = idx >> 15;                       // which matrix (32768 elems each)
  int r    = idx & 32767;
  int e    = r & 15;
  int lane = (r >> 4) & 31;
  int half = lane >> 4;
  int kk   = half * 16 + e;
  const float* src;
  int Kdim;
  if      (m == 0) { src = tw; Kdim = 256; }
  else if (m == 1) { src = pw; Kdim = 256; }
  else if (m == 2) { src = gw; Kdim = 256; }
  else             { src = ow; Kdim = 128; }
  int nk    = Kdim >> 5;
  int kstep = (r >> 9) % nk;
  int mtile = (r >> 9) / nk;
  int row   = mtile * 16 + (lane & 15);
  int col   = kstep * 32 + kk;
  wpack[idx] = f2bf(src[row * Kdim + col]);
}

// ---------- kernel 1: theta/phi/g projections ----------
// grid: 32 batches * 64 col-tiles; block 256 (8 waves). Wave w -> M rows [16w,16w+16).
__global__ __launch_bounds__(256) void proj_kernel(
    const float* __restrict__ x, const u16b* __restrict__ wpack,
    const float* __restrict__ tb, const float* __restrict__ pb,
    const float* __restrict__ gb, u16b* __restrict__ thetaB,
    u16b* __restrict__ phiB, u16b* __restrict__ gT) {
  __shared__ __attribute__((aligned(32))) u16b Xt[64 * 272];  // [n][c], padded
  const int b  = blockIdx.x >> 6;
  const int n0 = (blockIdx.x & 63) << 6;
  const int t  = threadIdx.x;
  const float* xb = x + (size_t)b * (C_IN * NSP);

#pragma unroll 4
  for (int i = 0; i < 64; ++i) {             // 16384 elems / 256 thr
    int idx = i * 256 + t;
    int c   = idx >> 6;
    int nn  = idx & 63;
    Xt[nn * 272 + c] = f2bf(xb[(size_t)c * NSP + n0 + nn]);
  }
  __syncthreads();

  const int lane = t & 31, wave = t >> 5;
  const int half = lane >> 4, l16 = lane & 15;
  const int mtile = wave;

  v8f accT[4], accP[4], accG[4];
#pragma unroll
  for (int nt = 0; nt < 4; ++nt) { accT[nt] = zero8(); accP[nt] = zero8(); accG[nt] = zero8(); }

#pragma unroll
  for (int ks = 0; ks < 8; ++ks) {
    const int aoff = ((mtile * 8 + ks) * 32 + lane) * 16;
    v16bf aT = frag_ld(wpack + aoff);           // theta_w packed
    v16bf aP = frag_ld(wpack + 32768 + aoff);   // phi_w packed
    v16bf aG = frag_ld(wpack + 65536 + aoff);   // g_w packed
#pragma unroll
    for (int nt = 0; nt < 4; ++nt) {
      v16bf bf = frag_ld(&Xt[(nt * 16 + l16) * 272 + ks * 32 + half * 16]);
      accT[nt] = wmma_bf16(aT, bf, accT[nt]);
      accP[nt] = wmma_bf16(aP, bf, accP[nt]);
      accG[nt] = wmma_bf16(aG, bf, accG[nt]);
    }
  }

  const int rbase = mtile * 16 + half * 8;
  float bT[8], bP[8], bG[8];
#pragma unroll
  for (int r = 0; r < 8; ++r) { bT[r] = tb[rbase + r]; bP[r] = pb[rbase + r]; bG[r] = gb[rbase + r]; }

  const size_t base = (size_t)b * (CI * NSP);
#pragma unroll
  for (int nt = 0; nt < 4; ++nt) {
    int col = n0 + nt * 16 + l16;
#pragma unroll
    for (int r = 0; r < 8; ++r) {
      int row = rbase + r;
      thetaB[base + (size_t)row * NSP + col] = f2bf(accT[nt][r] + bT[r]);
      phiB  [base + (size_t)row * NSP + col] = f2bf(accP[nt][r] + bP[r]);
    }
    v8us gv;  // g stored transposed [n][ci]: 8 consecutive rows -> 16B store
#pragma unroll
    for (int r = 0; r < 8; ++r) gv[r] = f2bf(accG[nt][r] + bG[r]);
    *(v8us*)(gT + base + (size_t)col * CI + rbase) = gv;
  }
}

// ---------- kernel 2: f = theta*phi^T, softmax rows, y^T = (f^T * g)^T ----------
// one workgroup per batch, 8 waves; wave = i-strip for f, j-strip for y.
__global__ __launch_bounds__(256) void attn_kernel(
    const u16b* __restrict__ thetaB, const u16b* __restrict__ phiB,
    const u16b* __restrict__ gT, u16b* __restrict__ yT) {
  __shared__ __attribute__((aligned(32))) u16b fT[128 * 136];  // [j][i] bf16
  const int b    = blockIdx.x;
  const int t    = threadIdx.x;
  const int lane = t & 31, wave = t >> 5;
  const int half = lane >> 4, l16 = lane & 15;
  const size_t base = (size_t)b * (CI * NSP);

  v8f acc[8];
#pragma unroll
  for (int nt = 0; nt < 8; ++nt) acc[nt] = zero8();

  const u16b* arow = thetaB + base + (size_t)(wave * 16 + l16) * NSP + half * 16;
  const u16b* prow = phiB + base + (size_t)l16 * NSP + half * 16;
  for (int ks = 0; ks < 128; ++ks) {          // K = 4096
    v16bf a = frag_cat(arow + ks * 32, arow + ks * 32 + 8);
#pragma unroll
    for (int nt = 0; nt < 8; ++nt) {
      v16bf bv = frag_ld(prow + (size_t)(nt * 16) * NSP + ks * 32);
      acc[nt] = wmma_bf16(a, bv, acc[nt]);
    }
  }

  // softmax over j for each row i (row lives in one 16-lane half x 8 accs)
#pragma unroll
  for (int r = 0; r < 8; ++r) {
    float m = acc[0][r];
#pragma unroll
    for (int nt = 1; nt < 8; ++nt) m = fmaxf(m, acc[nt][r]);
#pragma unroll
    for (int s = 1; s < 16; s <<= 1) m = fmaxf(m, __shfl_xor(m, s, 32));
    float sum = 0.0f;
#pragma unroll
    for (int nt = 0; nt < 8; ++nt) {
      float e = __expf(acc[nt][r] - m);
      acc[nt][r] = e;
      sum += e;
    }
#pragma unroll
    for (int s = 1; s < 16; s <<= 1) sum += __shfl_xor(sum, s, 32);
    float inv = 1.0f / sum;
    int i = wave * 16 + half * 8 + r;
#pragma unroll
    for (int nt = 0; nt < 8; ++nt)
      fT[(nt * 16 + l16) * 136 + i] = f2bf(acc[nt][r] * inv);
  }
  __syncthreads();

  // y[j][n] = sum_i fT[j][i] * g[i][n]; write transposed yT[n][j]
  v16bf af[4];
#pragma unroll
  for (int ks = 0; ks < 4; ++ks) {
    const u16b* p = &fT[(wave * 16 + l16) * 136 + ks * 32 + half * 16];
    af[ks] = frag_cat(p, p + 8);
  }
  const u16b* gbase = gT + base + (size_t)l16 * CI + half * 16;
  u16b* ybase = yT + base;
  for (int ch = 0; ch < 64; ++ch) {           // 4096 cols / (4*16)
    const int nb = ch * 64;
    v8f a2[4];
#pragma unroll
    for (int nt = 0; nt < 4; ++nt) a2[nt] = zero8();
#pragma unroll
    for (int ks = 0; ks < 4; ++ks) {
#pragma unroll
      for (int nt = 0; nt < 4; ++nt) {
        v16bf bv = frag_ld(gbase + (size_t)(nb + nt * 16) * CI + ks * 32);
        a2[nt] = wmma_bf16(af[ks], bv, a2[nt]);
      }
    }
#pragma unroll
    for (int nt = 0; nt < 4; ++nt) {
      int n = nb + nt * 16 + l16;
      v8us o;
#pragma unroll
      for (int r = 0; r < 8; ++r) o[r] = f2bf(a2[nt][r]);
      *(v8us*)(ybase + (size_t)n * CI + wave * 16 + half * 8) = o;
    }
  }
}

// ---------- kernel 3: out = out_w * Y + out_b + x ----------
// yT buffer reinterpreted per batch as row-major [128][4096] (the torch reshape).
__global__ __launch_bounds__(256) void out_kernel(
    const u16b* __restrict__ yT, const u16b* __restrict__ owpack,
    const float* __restrict__ ob, const float* __restrict__ x,
    float* __restrict__ out) {
  __shared__ __attribute__((aligned(32))) u16b Yl[64 * 144];  // [s][c'] padded
  const int b  = blockIdx.x >> 6;
  const int s0 = (blockIdx.x & 63) << 6;
  const int t  = threadIdx.x;
  const u16b* yb = yT + (size_t)b * (CI * NSP);

#pragma unroll 4
  for (int i = 0; i < 32; ++i) {              // 8192 elems / 256 thr
    int idx = i * 256 + t;
    int c   = idx >> 6;
    int ss  = idx & 63;
    Yl[ss * 144 + c] = yb[(size_t)c * NSP + s0 + ss];
  }
  __syncthreads();

  const int lane = t & 31, wave = t >> 5;
  const int half = lane >> 4, l16 = lane & 15;

  v8f acc[2][4];
#pragma unroll
  for (int mm = 0; mm < 2; ++mm)
#pragma unroll
    for (int nt = 0; nt < 4; ++nt) acc[mm][nt] = zero8();

#pragma unroll
  for (int ks = 0; ks < 4; ++ks) {            // K = 128
    v16bf a0 = frag_ld(owpack + ((wave * 4 + ks) * 32 + lane) * 16);
    v16bf a1 = frag_ld(owpack + (((wave + 8) * 4 + ks) * 32 + lane) * 16);
#pragma unroll
    for (int nt = 0; nt < 4; ++nt) {
      v16bf bv = frag_ld(&Yl[(nt * 16 + l16) * 144 + ks * 32 + half * 16]);
      acc[0][nt] = wmma_bf16(a0, bv, acc[0][nt]);
      acc[1][nt] = wmma_bf16(a1, bv, acc[1][nt]);
    }
  }

  const size_t obase = (size_t)b * (size_t)(C_IN * NSP);
#pragma unroll
  for (int mm = 0; mm < 2; ++mm) {
    const int rbase = (wave + mm * 8) * 16 + half * 8;
    float bias[8];
#pragma unroll
    for (int r = 0; r < 8; ++r) bias[r] = ob[rbase + r];
#pragma unroll
    for (int nt = 0; nt < 4; ++nt) {
      int s = s0 + nt * 16 + l16;
#pragma unroll
      for (int r = 0; r < 8; ++r) {
        size_t oi = obase + (size_t)(rbase + r) * NSP + s;
        out[oi] = acc[mm][nt][r] + bias[r] + x[oi];
      }
    }
  }
}

// ---------- launch ----------
extern "C" void kernel_launch(void* const* d_in, const int* in_sizes, int n_in,
                              void* d_out, int out_size, void* d_ws, size_t ws_size,
                              hipStream_t stream) {
  const float* x  = (const float*)d_in[0];
  const float* tw = (const float*)d_in[1];
  const float* tb = (const float*)d_in[2];
  const float* pw = (const float*)d_in[3];
  const float* pb = (const float*)d_in[4];
  const float* gw = (const float*)d_in[5];
  const float* gb = (const float*)d_in[6];
  const float* ow = (const float*)d_in[7];
  const float* ob = (const float*)d_in[8];

  u16b* wp     = (u16b*)d_ws;                 // 4 packed weight mats: 131072 elems
  u16b* thetaB = wp + 131072;                 // [B][CI][N] bf16
  u16b* phiB   = thetaB + (size_t)BATCH * CI * NSP;
  u16b* gT     = phiB   + (size_t)BATCH * CI * NSP;  // [B][N][CI] bf16
  u16b* yT     = gT     + (size_t)BATCH * CI * NSP;  // [B][N][CI] bf16 (== torch reshape)

  pack_weights_kernel<<<512, 256, 0, stream>>>(tw, pw, gw, ow, wp);
  proj_kernel<<<BATCH * 64, 256, 0, stream>>>(x, wp, tb, pb, gb, thetaB, phiB, gT);
  attn_kernel<<<BATCH, 256, 0, stream>>>(thetaB, phiB, gT, yT);
  out_kernel<<<BATCH * 64, 256, 0, stream>>>(yT, wp + 98304, ob, x, (float*)d_out);
}
